// InundationCoder_41317585387565
// MI455X (gfx1250) — matrix operand order repaired and checked by reading, not compiled, not run on hardware
//
#include <hip/hip_runtime.h>

// ---------------- model constants (match reference) ----------------
constexpr int N_ = 8192, B_ = 32, T_ = 32, E_ = 32768;
constexpr int NT_ = N_ * T_;        // 262144 rows
constexpr int EP_ = E_ + N_;        // edges + self loops = 40960
constexpr int LH_ = 128;            // LSTM hidden

typedef __attribute__((ext_vector_type(16))) _Float16 v16h;
typedef __attribute__((ext_vector_type(8)))  _Float16 v8h;
typedef __attribute__((ext_vector_type(8)))  float    v8f;

// order-preserving float<->uint encoding for atomicMax on floats
__device__ __forceinline__ unsigned fenc(float f) {
  unsigned u = __float_as_uint(f);
  return (u & 0x80000000u) ? ~u : (u | 0x80000000u);
}
__device__ __forceinline__ float fdec(unsigned u) {
  return (u & 0x80000000u) ? __uint_as_float(u & 0x7FFFFFFFu) : __uint_as_float(~u);
}
__device__ __forceinline__ float sigmoidf_(float x) { return 1.f / (1.f + __expf(-x)); }

__device__ __forceinline__ v16h cat8(v8h a0, v8h a1) {
  return __builtin_shufflevector(a0, a1, 0, 1, 2, 3, 4, 5, 6, 7, 8, 9, 10, 11, 12, 13, 14, 15);
}

// ---------------- weight swizzle into B-fragment order ----------------
// dst[(kt*(N/16)+nt)*512 + lane*16 + i] = src[(kt*32 + (lane>>4)*8 + (i&7) + (i>>3)*16) * N + nt*16 + (lane&15)]
// rows [0,rows0) from s0, [rows0,rows0+rows1) from s1, rest zero-padded.
__global__ void k_pack_swz(const float* __restrict__ s0, int rows0,
                           const float* __restrict__ s1, int rows1,
                           _Float16* __restrict__ dst, int Ncols, int total) {
  int idx = blockIdx.x * 256 + threadIdx.x;
  if (idx >= total) return;
  int tile = idx >> 9, r = idx & 511;
  int lane = r >> 4, i = r & 15;
  int ntiles = Ncols >> 4;
  int kt = tile / ntiles, nt = tile % ntiles;
  int k = kt * 32 + (lane >> 4) * 8 + (i & 7) + (i >> 3) * 16;
  int col = nt * 16 + (lane & 15);
  float v = 0.f;
  if (k < rows0)              v = s0[(size_t)k * Ncols + col];
  else if (k < rows0 + rows1) v = s1[(size_t)(k - rows0) * Ncols + col];
  dst[idx] = (_Float16)v;
}

// packed input rows [NT, 64] f16 = [era5 | bc | bd | 0], coalesced
__global__ void k_pack_xin(const float* __restrict__ era5, const float* __restrict__ bc,
                           const float* __restrict__ bd, _Float16* __restrict__ X) {
  int idx = blockIdx.x * 256 + threadIdx.x;      // NT*64
  if (idx >= NT_ * 64) return;
  int r = idx >> 6, k = idx & 63, node = r >> 5; // T == 32
  float v;
  if (k < 16)      v = era5[(size_t)r * 16 + k];
  else if (k < 32) v = bc[node * 16 + (k - 16)];
  else if (k < 40) v = bd[node * 8 + (k - 32)];
  else             v = 0.f;
  X[idx] = (_Float16)v;
}

// packed river features [B, 32] f16 = [rc | rd | 0]
__global__ void k_pack_rcrd(const float* __restrict__ rc, const float* __restrict__ rd,
                            _Float16* __restrict__ X) {
  int idx = blockIdx.x * 256 + threadIdx.x;      // B*32
  if (idx >= B_ * 32) return;
  int b = idx >> 5, k = idx & 31;
  float v = (k < 16) ? rc[b * 16 + k] : (k < 24) ? rd[b * 8 + (k - 16)] : 0.f;
  X[idx] = (_Float16)v;
}

__global__ void k_batchidx(const int* __restrict__ nodes, int* __restrict__ bidx) {
  int b = threadIdx.x;
  if (b < B_) { int s = 0; for (int i = 0; i < b; i++) s += nodes[i]; bidx[b] = s; }
}
__global__ void k_fill_f32(float* __restrict__ p, float v, int n) {
  int i = blockIdx.x * 256 + threadIdx.x;
  if (i < n) p[i] = v;
}
__global__ void k_fill_u32(unsigned* __restrict__ p, unsigned v, int n) {
  int i = blockIdx.x * 256 + threadIdx.x;
  if (i < n) p[i] = v;
}

// ---------------- WMMA GEMM stripe core: one wave does 16 rows x all 64 cols ----------------
// A fragments are loaded once per k-step and reused across the 4 column tiles
// (4x less activation HBM traffic; weights are L2-resident).
__device__ __forceinline__ void gemm64_stripe(const _Float16* __restrict__ A,
                                              const _Float16* __restrict__ Wswz,
                                              int rowTile, int lane, v8f c[4]) {
  int m = lane & 15, hi = lane >> 4;
  size_t rbase = (size_t)(rowTile * 16 + m) * 64;
#pragma unroll
  for (int kt = 0; kt < 2; kt++) {
    v8h a0 = *(const v8h*)(A + rbase + kt * 32 + hi * 8);        // global_load_b128
    v8h a1 = *(const v8h*)(A + rbase + kt * 32 + 16 + hi * 8);   // global_load_b128
    v16h a = cat8(a0, a1);
#pragma unroll
    for (int ct = 0; ct < 4; ct++) {
      v16h b = *(const v16h*)(Wswz + (size_t)(kt * 4 + ct) * 512 + lane * 16);
      c[ct] = __builtin_amdgcn_wmma_f32_16x16x32_f16(false, a, false, b, (short)0, c[ct], false, false);
    }
  }
}

// out_f16 = relu(A @ W + bias)   (input projection)
__global__ void k_gemm_bias_relu_f16(const _Float16* __restrict__ A,
                                     const _Float16* __restrict__ Wswz,
                                     const float* __restrict__ bias,
                                     _Float16* __restrict__ outp) {
  int lane = threadIdx.x & 31, wave = threadIdx.x >> 5;
  int rowTile = blockIdx.x * 8 + wave;           // NT/16 stripes
  int hi = lane >> 4;
  v8f c[4] = {v8f{}, v8f{}, v8f{}, v8f{}};
  gemm64_stripe(A, Wswz, rowTile, lane, c);
  int mbase = rowTile * 16 + hi * 8;
#pragma unroll
  for (int ct = 0; ct < 4; ct++) {
    int col = ct * 16 + (lane & 15);
    float bv = bias[col];
#pragma unroll
    for (int v = 0; v < 8; v++)
      outp[(size_t)(mbase + v) * 64 + col] = (_Float16)fmaxf(c[ct][v] + bv, 0.f);
  }
}

// out_f32 = A @ W   (GAT h projection; f32 needed for attention + aggregation)
__global__ void k_gemm_f32(const _Float16* __restrict__ A, const _Float16* __restrict__ Wswz,
                           float* __restrict__ outp) {
  int lane = threadIdx.x & 31, wave = threadIdx.x >> 5;
  int rowTile = blockIdx.x * 8 + wave;
  int hi = lane >> 4;
  v8f c[4] = {v8f{}, v8f{}, v8f{}, v8f{}};
  gemm64_stripe(A, Wswz, rowTile, lane, c);
  int mbase = rowTile * 16 + hi * 8;
#pragma unroll
  for (int ct = 0; ct < 4; ct++) {
    int col = ct * 16 + (lane & 15);
#pragma unroll
    for (int v = 0; v < 8; v++)
      outp[(size_t)(mbase + v) * 64 + col] = c[ct][v];
  }
}

// ---------------- attention logits a_s/a_d per (row, head) ----------------
__global__ void k_asad(const float* __restrict__ h, const float* __restrict__ asrc,
                       const float* __restrict__ adst, float* __restrict__ aS,
                       float* __restrict__ aD) {
  int idx = blockIdx.x * 256 + threadIdx.x;      // NT*4
  if (idx >= NT_ * 4) return;
  int row = idx >> 2, hh = idx & 3;
  const float* hp = h + (size_t)row * 64 + hh * 16;
  float s = 0.f, d = 0.f;
#pragma unroll
  for (int k = 0; k < 16; k++) {
    s += hp[k] * asrc[hh * 16 + k];
    d += hp[k] * adst[hh * 16 + k];
  }
  aS[idx] = s; aD[idx] = d;
}

// ---------------- edge softmax: max, exp+den, aggregate ----------------
__global__ void k_edge_max(const int* __restrict__ edge, const float* __restrict__ aS,
                           const float* __restrict__ aD, unsigned* __restrict__ menc) {
  int idx = blockIdx.x * 256 + threadIdx.x;      // EP*T*4
  if (idx >= EP_ * T_ * 4) return;
  int e = idx >> 7, rem = idx & 127, t = rem >> 2, hh = rem & 3;
  int s = (e < E_) ? edge[e] : (e - E_);
  int d = (e < E_) ? edge[E_ + e] : (e - E_);
  float v = aS[((size_t)s * T_ + t) * 4 + hh] + aD[((size_t)d * T_ + t) * 4 + hh];
  float lk = v > 0.f ? v : 0.2f * v;
  atomicMax(&menc[((size_t)d * T_ + t) * 4 + hh], fenc(lk));
}
__global__ void k_edge_w(const int* __restrict__ edge, const float* __restrict__ aS,
                         const float* __restrict__ aD, const unsigned* __restrict__ menc,
                         float* __restrict__ wE, float* __restrict__ den) {
  int idx = blockIdx.x * 256 + threadIdx.x;      // EP*T*4
  if (idx >= EP_ * T_ * 4) return;
  int e = idx >> 7, rem = idx & 127, t = rem >> 2, hh = rem & 3;
  int s = (e < E_) ? edge[e] : (e - E_);
  int d = (e < E_) ? edge[E_ + e] : (e - E_);
  float v = aS[((size_t)s * T_ + t) * 4 + hh] + aD[((size_t)d * T_ + t) * 4 + hh];
  float lk = v > 0.f ? v : 0.2f * v;
  float w = __expf(lk - fdec(menc[((size_t)d * T_ + t) * 4 + hh]));
  wE[idx] = w;
  atomicAdd(&den[((size_t)d * T_ + t) * 4 + hh], w);
}
__global__ void k_edge_agg(const int* __restrict__ edge, const float* __restrict__ wE,
                           const float* __restrict__ den, const float* __restrict__ hsrc,
                           float* __restrict__ outp) {
  int idx = blockIdx.x * 256 + threadIdx.x;      // EP*T
  if (idx >= EP_ * T_) return;
  int e = idx >> 5, t = idx & 31;
  int s = (e < E_) ? edge[e] : (e - E_);
  int d = (e < E_) ? edge[E_ + e] : (e - E_);
  float al[4];
#pragma unroll
  for (int hh = 0; hh < 4; hh++)
    al[hh] = wE[(size_t)e * 128 + t * 4 + hh] /
             (den[((size_t)d * T_ + t) * 4 + hh] + 1e-16f);
  const float* hs = hsrc + ((size_t)s * T_ + t) * 64;
  float* op = outp + ((size_t)d * T_ + t) * 64;
#pragma unroll 4
  for (int k = 0; k < 64; k++) atomicAdd(&op[k], al[k >> 4] * hs[k]);
}

// acc(f32) + bias [-> relu] -> X (f16), feeds the next WMMA GEMM
__global__ void k_bias_act_f16(const float* __restrict__ acc, const float* __restrict__ bias,
                               _Float16* __restrict__ X, int relu) {
  int i = blockIdx.x * 256 + threadIdx.x;        // NT*64
  if (i >= NT_ * 64) return;
  float v = acc[i] + bias[i & 63];
  if (relu) v = fmaxf(v, 0.f);
  X[i] = (_Float16)v;
}

// ---------------- river proj: relu([sampled,rc,rd]@Wr + b_r), rows=B*T, K=96 ----------------
__global__ void k_river_proj(const _Float16* __restrict__ Xb, const int* __restrict__ bidx,
                             const _Float16* __restrict__ rcrd, const _Float16* __restrict__ Wswz,
                             const float* __restrict__ bias, _Float16* __restrict__ series) {
  int lane = threadIdx.x & 31, wave = threadIdx.x >> 5;
  int rowTile = blockIdx.x * 8 + wave;           // 64 row stripes (B*T/16)
  int m = lane & 15, hi = lane >> 4;
  int r = rowTile * 16 + m;                      // r = b*T + t
  int b = r >> 5, t = r & 31;
  int node = bidx[b];
  size_t abase = ((size_t)node * T_ + t) * 64;
  v8f c[4] = {v8f{}, v8f{}, v8f{}, v8f{}};
#pragma unroll
  for (int kt = 0; kt < 3; kt++) {
    v8h a0, a1;
    if (kt < 2) {                                // sampled attention features
      a0 = *(const v8h*)(Xb + abase + kt * 32 + hi * 8);
      a1 = *(const v8h*)(Xb + abase + kt * 32 + 16 + hi * 8);
    } else {                                     // pre-packed river features
      a0 = *(const v8h*)(rcrd + b * 32 + hi * 8);
      a1 = *(const v8h*)(rcrd + b * 32 + 16 + hi * 8);
    }
    v16h a = cat8(a0, a1);
#pragma unroll
    for (int ct = 0; ct < 4; ct++) {
      v16h bf = *(const v16h*)(Wswz + (size_t)(kt * 4 + ct) * 512 + lane * 16);
      c[ct] = __builtin_amdgcn_wmma_f32_16x16x32_f16(false, a, false, bf, (short)0, c[ct], false, false);
    }
  }
  int mbase = rowTile * 16 + hi * 8;
#pragma unroll
  for (int ct = 0; ct < 4; ct++) {
    int col = ct * 16 + (lane & 15);
    float bv = bias[col];
#pragma unroll
    for (int v = 0; v < 8; v++)
      series[(size_t)(mbase + v) * 64 + col] = (_Float16)fmaxf(c[ct][v] + bv, 0.f);
  }
}

// ---------------- persistent single-block WMMA LSTM ----------------
// G = [x_t ; h] @ [Wi ; Wh] + b with the combined weight pre-swizzled ([192,512]).
__global__ void k_lstm(const _Float16* __restrict__ series, const _Float16* __restrict__ Wswz,
                       const float* __restrict__ bias, float* __restrict__ seq,
                       float* __restrict__ hT, float* __restrict__ cT) {
  __shared__ _Float16 Gs[B_ * 512];    // 32 KB
  __shared__ _Float16 Hs[B_ * LH_];    //  8 KB
  __shared__ float    Cs[B_ * LH_];    // 16 KB
  int tid = threadIdx.x, lane = tid & 31, wave = tid >> 5;
  for (int i = tid; i < B_ * LH_; i += 256) { Hs[i] = (_Float16)0.f; Cs[i] = 0.f; }
  __syncthreads();
  for (int t = 0; t < T_; t++) {
    for (int tile = wave; tile < 64; tile += 8) {
      int mt = tile >> 5, nt = tile & 31;
      int m = lane & 15, hi = lane >> 4;
      int row = mt * 16 + m;                     // batch index
      int col = nt * 16 + (lane & 15);
      size_t sbase = ((size_t)row * T_ + t) * 64;
      v8f c = {};
#pragma unroll
      for (int kt = 0; kt < 2; kt++) {           // x_t part (global f16)
        v8h a0 = *(const v8h*)(series + sbase + kt * 32 + hi * 8);
        v8h a1 = *(const v8h*)(series + sbase + kt * 32 + 16 + hi * 8);
        v16h a = cat8(a0, a1);
        v16h bf = *(const v16h*)(Wswz + (size_t)(kt * 32 + nt) * 512 + lane * 16);
        c = __builtin_amdgcn_wmma_f32_16x16x32_f16(false, a, false, bf, (short)0, c, false, false);
      }
#pragma unroll
      for (int kt = 2; kt < 6; kt++) {           // h part (LDS f16, ds_load_b128)
        const _Float16* hp = Hs + row * LH_ + (kt - 2) * 32;
        v8h a0 = *(const v8h*)(hp + hi * 8);
        v8h a1 = *(const v8h*)(hp + 16 + hi * 8);
        v16h a = cat8(a0, a1);
        v16h bf = *(const v16h*)(Wswz + (size_t)(kt * 32 + nt) * 512 + lane * 16);
        c = __builtin_amdgcn_wmma_f32_16x16x32_f16(false, a, false, bf, (short)0, c, false, false);
      }
      int mbase = mt * 16 + hi * 8;
      float bj = bias[col];
#pragma unroll
      for (int v = 0; v < 8; v++) Gs[(mbase + v) * 512 + col] = (_Float16)(c[v] + bj);
    }
    __syncthreads();
    for (int i = tid; i < B_ * LH_; i += 256) {  // gate phase (torch order i,f,g,o)
      int b = i >> 7, j = i & 127;
      float gi = (float)Gs[b * 512 + j];
      float gf = (float)Gs[b * 512 + 128 + j];
      float gg = (float)Gs[b * 512 + 256 + j];
      float go = (float)Gs[b * 512 + 384 + j];
      float cc = sigmoidf_(gf) * Cs[i] + sigmoidf_(gi) * tanhf(gg);
      float hh = sigmoidf_(go) * tanhf(cc);
      Cs[i] = cc; Hs[i] = (_Float16)hh;
      seq[((size_t)b * T_ + t) * LH_ + j] = hh;
      if (t == T_ - 1) { hT[i] = hh; cT[i] = cc; }
    }
    __syncthreads();
  }
}

// ---------------- CMAL head ----------------
__global__ void k_head(const float* __restrict__ seq, const float* __restrict__ W,
                       const float* __restrict__ bh, float* __restrict__ cast) {
  int r = blockIdx.x * 256 + threadIdx.x;        // B*T = 1024 rows
  if (r >= B_ * T_) return;
  float z[12];
#pragma unroll
  for (int j = 0; j < 12; j++) z[j] = bh[j];
  const float* s = seq + (size_t)r * LH_;
  for (int k = 0; k < LH_; k++) {
    float sv = s[k];
    const float* w = W + k * 12;
#pragma unroll
    for (int j = 0; j < 12; j++) z[j] += sv * w[j];
  }
  float o[12];
#pragma unroll
  for (int j = 0; j < 3; j++) o[j] = z[j];
#pragma unroll
  for (int j = 0; j < 3; j++) {
    float x = z[3 + j];
    o[3 + j] = ((x > 20.f) ? x : log1pf(__expf(x))) + 1e-3f;
  }
#pragma unroll
  for (int j = 0; j < 3; j++) o[6 + j] = sigmoidf_(z[6 + j]);
  float mx = fmaxf(z[9], fmaxf(z[10], z[11]));
  float e0 = __expf(z[9] - mx), e1 = __expf(z[10] - mx), e2 = __expf(z[11] - mx);
  float sm = e0 + e1 + e2;
  o[9] = e0 / sm; o[10] = e1 / sm; o[11] = e2 / sm;
#pragma unroll
  for (int j = 0; j < 12; j++) cast[(size_t)r * 12 + j] = o[j];
}

// ---------------- driver ----------------
extern "C" void kernel_launch(void* const* d_in, const int* in_sizes, int n_in,
                              void* d_out, int out_size, void* d_ws, size_t ws_size,
                              hipStream_t stream) {
  (void)in_sizes; (void)n_in; (void)out_size; (void)ws_size;
  const float* era5 = (const float*)d_in[0];
  const float* bc   = (const float*)d_in[1];
  const float* bd   = (const float*)d_in[2];
  const float* rc   = (const float*)d_in[3];
  const float* rd   = (const float*)d_in[4];
  const float* Wc_b = (const float*)d_in[5];
  const float* Wd_b = (const float*)d_in[6];
  const float* b_b  = (const float*)d_in[7];
  const float* gW1  = (const float*)d_in[8];
  const float* as1  = (const float*)d_in[9];
  const float* ad1  = (const float*)d_in[10];
  const float* gb1  = (const float*)d_in[11];
  const float* gW2  = (const float*)d_in[12];
  const float* as2  = (const float*)d_in[13];
  const float* ad2  = (const float*)d_in[14];
  const float* gb2  = (const float*)d_in[15];
  const float* Wc_r = (const float*)d_in[16];
  const float* Wd_r = (const float*)d_in[17];
  const float* b_r  = (const float*)d_in[18];
  const float* Wi   = (const float*)d_in[19];
  const float* Wh   = (const float*)d_in[20];
  const float* lb   = (const float*)d_in[21];
  const float* hW   = (const float*)d_in[22];
  const float* hb   = (const float*)d_in[23];
  const int*   edge = (const int*)d_in[24];
  const int*   nodes= (const int*)d_in[25];
  float* out = (float*)d_out;

  char* ws = (char*)d_ws;
  size_t off = 0;
  auto alloc = [&](size_t bytes) { size_t p = off; off += (bytes + 255) & ~(size_t)255; return p; };
  _Float16* Xa    = (_Float16*)(ws + alloc((size_t)NT_ * 64 * 2));  // packed input / act ping
  _Float16* Xb    = (_Float16*)(ws + alloc((size_t)NT_ * 64 * 2));  // act pong
  float*    Hf32  = (float*)(ws + alloc((size_t)NT_ * 64 * 4));     // GAT h (f32)
  float*    Acc   = (float*)(ws + alloc((size_t)NT_ * 64 * 4));     // atomic aggregation
  float*    aS    = (float*)(ws + alloc((size_t)NT_ * 4 * 4));
  float*    aD    = (float*)(ws + alloc((size_t)NT_ * 4 * 4));
  unsigned* mEnc  = (unsigned*)(ws + alloc((size_t)NT_ * 4 * 4));
  float*    den   = (float*)(ws + alloc((size_t)NT_ * 4 * 4));
  float*    wE    = (float*)(ws + alloc((size_t)EP_ * T_ * 4 * 4));
  _Float16* rcrd  = (_Float16*)(ws + alloc(B_ * 32 * 2));
  _Float16* series= (_Float16*)(ws + alloc((size_t)B_ * T_ * 64 * 2));
  float*    seq   = (float*)(ws + alloc((size_t)B_ * T_ * LH_ * 4));
  int*      bidx  = (int*)(ws + alloc(B_ * 4));
  _Float16* WbS   = (_Float16*)(ws + alloc(64 * 64 * 2));
  _Float16* W1S   = (_Float16*)(ws + alloc(64 * 64 * 2));
  _Float16* W2S   = (_Float16*)(ws + alloc(64 * 64 * 2));
  _Float16* WrS   = (_Float16*)(ws + alloc(96 * 64 * 2));
  _Float16* WihS  = (_Float16*)(ws + alloc(192 * 512 * 2));

  const unsigned NEG_INF_ENC = 0x007FFFFFu;   // fenc(-inf)
  // -------- weight prep (swizzled to B-fragment order) --------
  k_pack_swz<<<16, 256, 0, stream>>>(Wc_b, 32, Wd_b, 8, WbS, 64, 64 * 64);
  k_pack_swz<<<16, 256, 0, stream>>>(gW1, 64, nullptr, 0, W1S, 64, 64 * 64);
  k_pack_swz<<<16, 256, 0, stream>>>(gW2, 64, nullptr, 0, W2S, 64, 64 * 64);
  k_pack_swz<<<24, 256, 0, stream>>>(Wc_r, 80, Wd_r, 8, WrS, 64, 96 * 64);
  k_pack_swz<<<384, 256, 0, stream>>>(Wi, 64, Wh, 128, WihS, 512, 192 * 512);
  k_batchidx<<<1, 32, 0, stream>>>(nodes, bidx);
  k_pack_rcrd<<<4, 256, 0, stream>>>(rc, rd, rcrd);

  const int stripeBlocks = NT_ / 16 / 8;        // 2048 (one 16x64 stripe per wave)
  const int nth4 = NT_ * 4;                     // 1M
  const int eth  = EP_ * T_ * 4;                // 5.24M

  // -------- input projection --------
  k_pack_xin<<<NT_ * 64 / 256, 256, 0, stream>>>(era5, bc, bd, Xa);
  k_gemm_bias_relu_f16<<<stripeBlocks, 256, 0, stream>>>(Xa, WbS, b_b, Xb);

  // -------- GAT layer 1 --------
  k_gemm_f32<<<stripeBlocks, 256, 0, stream>>>(Xb, W1S, Hf32);          // h1
  k_asad<<<nth4 / 256, 256, 0, stream>>>(Hf32, as1, ad1, aS, aD);
  k_fill_u32<<<nth4 / 256, 256, 0, stream>>>(mEnc, NEG_INF_ENC, nth4);
  k_fill_f32<<<nth4 / 256, 256, 0, stream>>>(den, 0.f, nth4);
  k_edge_max<<<(eth + 255) / 256, 256, 0, stream>>>(edge, aS, aD, mEnc);
  k_edge_w<<<(eth + 255) / 256, 256, 0, stream>>>(edge, aS, aD, mEnc, wE, den);
  k_fill_f32<<<NT_ * 64 / 256, 256, 0, stream>>>(Acc, 0.f, NT_ * 64);
  k_edge_agg<<<(EP_ * T_ + 255) / 256, 256, 0, stream>>>(edge, wE, den, Hf32, Acc);
  k_bias_act_f16<<<NT_ * 64 / 256, 256, 0, stream>>>(Acc, gb1, Xa, 1);  // relu -> Xa

  // -------- GAT layer 2 --------
  k_gemm_f32<<<stripeBlocks, 256, 0, stream>>>(Xa, W2S, Hf32);          // h2
  k_asad<<<nth4 / 256, 256, 0, stream>>>(Hf32, as2, ad2, aS, aD);
  k_fill_u32<<<nth4 / 256, 256, 0, stream>>>(mEnc, NEG_INF_ENC, nth4);
  k_fill_f32<<<nth4 / 256, 256, 0, stream>>>(den, 0.f, nth4);
  k_edge_max<<<(eth + 255) / 256, 256, 0, stream>>>(edge, aS, aD, mEnc);
  k_edge_w<<<(eth + 255) / 256, 256, 0, stream>>>(edge, aS, aD, mEnc, wE, den);
  k_fill_f32<<<NT_ * 64 / 256, 256, 0, stream>>>(Acc, 0.f, NT_ * 64);
  k_edge_agg<<<(EP_ * T_ + 255) / 256, 256, 0, stream>>>(edge, wE, den, Hf32, Acc);
  k_bias_act_f16<<<NT_ * 64 / 256, 256, 0, stream>>>(Acc, gb2, Xb, 0);  // no relu -> Xb

  // -------- sample gauges + river projection --------
  k_river_proj<<<8, 256, 0, stream>>>(Xb, bidx, rcrd, WrS, b_r, series);

  // -------- LSTM (persistent single block) --------
  k_lstm<<<1, 256, 0, stream>>>(series, WihS, lb, seq,
                                out + B_ * T_ * 12, out + B_ * T_ * 12 + B_ * LH_);

  // -------- CMAL head --------
  k_head<<<4, 256, 0, stream>>>(seq, hW, hb, out);
}
